// IGConv_71322226917424
// MI455X (gfx1250) — matrix operand order repaired and checked by reading, not compiled, not run on hardware
//
#include <hip/hip_runtime.h>
#include <hip/hip_bf16.h>

// ---------------------------------------------------------------------------
// Problem constants (from the reference)
// ---------------------------------------------------------------------------
#define BS    32
#define NA    128
#define NE    16256          // NA*(NA-1)
#define FEAT  32
#define EMB   64
#define CONV_OUT 256         // CC*NU*D
#define MLP1_IN  320
#define MLP2_IN  160
#define H1    256
#define LFPAD 328            // LDS row stride (bf16 elems), >= 320, even
#define WAVES 2              // waves per block
#define MT    2              // 16-row M tiles per wave (32 rows / wave)
#define MROWS (MT * 16)

typedef __bf16 bf16;
typedef __attribute__((ext_vector_type(16))) __bf16 v16bf;
typedef __attribute__((ext_vector_type(2)))  __bf16 v2bf;
typedef __attribute__((ext_vector_type(8)))  float  v8f;
typedef __attribute__((ext_vector_type(4)))  float  v4f;

// ---------------------------------------------------------------------------
// WMMA fragment layouts (wave32, v_wmma_f32_16x16x32_bf16)
//
// A (16x32, MxK): lane l -> row m = l&15, half h = l>>4.
//   VGPR v holds K = (v<4 ? 0:16) + h*8 + (v&3)*2  (+1 in high bf16).
// B (32x16, KxN): lane l -> col n = l&15.
//   element j (0..15) of the 16-bf16 lane vector holds K = (j>>3)*16 + (l>>4)*8 + (j&7).
// C/D (16x16, f32): lane l -> col n = l&15; VGPR v -> row m = (l>>4)*8 + v.
// ---------------------------------------------------------------------------

// Load an A-fragment for LDS row `row` (row-major bf16, stride LFPAD),
// K window [kf*32, kf*32+32).
__device__ __forceinline__ v16bf load_a_frag(const bf16 (&tile)[MROWS][LFPAD],
                                             int row, int h, int kf) {
  v16bf a;
#pragma unroll
  for (int v = 0; v < 8; ++v) {
    const int k = kf * 32 + ((v < 4) ? 0 : 16) + h * 8 + (v & 3) * 2;
    v2bf p = *(const v2bf*)&tile[row][k];   // 4B-aligned ds_load_b32
    a[2 * v]     = p[0];
    a[2 * v + 1] = p[1];
  }
  return a;
}

// ---------------------------------------------------------------------------
// Weight prepack: f32 (K,N) row-major -> per-lane B-fragment layout.
// out[((kf*(N/16)+nt)*32 + lane)*16 + j] = bf16(W[k*N + n])
//   n = nt*16 + (lane&15),  k = kf*32 + ((j>>3)<<4) + ((lane>>4)<<3) + (j&7)
// ---------------------------------------------------------------------------
__global__ void pack_b_kernel(const float* __restrict__ W, bf16* __restrict__ out,
                              int K, int N) {
  const int t = blockIdx.x * blockDim.x + threadIdx.x;
  if (t >= K * N) return;
  const int j    = t & 15;
  const int lane = (t >> 4) & 31;
  const int fi   = t >> 9;
  const int ntc  = N >> 4;
  const int kf   = fi / ntc;
  const int nt   = fi - kf * ntc;
  const int n    = nt * 16 + (lane & 15);
  const int k    = kf * 32 + ((j >> 3) << 4) + ((lane >> 4) << 3) + (j & 7);
  out[t] = (bf16)W[k * N + n];
}

__global__ void zero_kernel(float* __restrict__ p, int n) {
  const int i = blockIdx.x * blockDim.x + threadIdx.x;
  if (i < n) p[i] = 0.f;
}

// ---------------------------------------------------------------------------
// Edge kernel: conv+ReLU -> MLP1 (320->256 ReLU -> 64) -> scatter-add to agg.
// Each wave owns a 32-edge tile (two 16-row WMMA tiles sharing B fragments).
// ---------------------------------------------------------------------------
__global__ __launch_bounds__(32 * WAVES, 1)
void edge_kernel(const float* __restrict__ x0,
                 const float* __restrict__ x_last,
                 const float* __restrict__ edge_attr,
                 const float* __restrict__ conv_w,
                 const float* __restrict__ conv_b,
                 const float* __restrict__ b1,
                 const float* __restrict__ b2,
                 const int*   __restrict__ edge_idx,
                 const bf16*  __restrict__ W1b,
                 const bf16*  __restrict__ W2b,
                 float*       __restrict__ agg) {
  __shared__ bf16 smem[WAVES][MROWS][LFPAD];

  const int tid  = threadIdx.x;
  const int lane = tid & 31;
  const int wave = tid >> 5;
  const int b    = blockIdx.y;
  const int e0   = (blockIdx.x * WAVES + wave) * MROWS;
  const int m    = lane & 15;   // row-in-tile / output column
  const int h    = lane >> 4;   // half-wave id
  bf16 (&lf)[MROWS][LFPAD] = smem[wave];

  // ---- stage 1: gather + fused conv(ReLU) + x_last -> lf[32][320] bf16 ----
#pragma unroll
  for (int r = 0; r < MT; ++r) {
    const int row = r * 16 + m;
    const int e   = e0 + row;
    const int src = edge_idx[2 * e];
    const float* nfp = x0        + ((size_t)b * NA + src) * FEAT + h * 16;
    const float* eap = edge_attr + ((size_t)b * NE + e)   * FEAT + h * 16;
    float nf[16], ea[16];
#pragma unroll
    for (int i = 0; i < 4; ++i) {
      *(v4f*)&nf[4 * i] = *(const v4f*)&nfp[4 * i];
      *(v4f*)&ea[4 * i] = *(const v4f*)&eap[4 * i];
    }
#pragma unroll
    for (int o = 0; o < 8; ++o) {
      const float w0 = conv_w[2 * o], w1 = conv_w[2 * o + 1], cb = conv_b[o];
#pragma unroll
      for (int j = 0; j < 16; ++j) {
        const float y = fmaxf(w0 * nf[j] + w1 * ea[j] + cb, 0.f);
        lf[row][o * 32 + h * 16 + j] = (bf16)y;
      }
    }
    const float* lsp = x_last + ((size_t)b * NA + src) * EMB + h * 32;
#pragma unroll
    for (int j = 0; j < 32; ++j)
      lf[row][CONV_OUT + h * 32 + j] = (bf16)lsp[j];
  }
  __syncthreads();

  // ---- preload biases (keeps the GEMM loops free of global loads) ----
  float bias1[16], bias2[4];
#pragma unroll
  for (int nt = 0; nt < 16; ++nt) bias1[nt] = b1[nt * 16 + m];
#pragma unroll
  for (int nt = 0; nt < 4; ++nt)  bias2[nt] = b2[nt * 16 + m];

  // ---- A-fragments for GEMM1 (K = 320 -> 10 frags x MT tiles) ----
  v16bf afrag[MT][10];
#pragma unroll
  for (int t = 0; t < MT; ++t)
#pragma unroll
    for (int kf = 0; kf < 10; ++kf)
      afrag[t][kf] = load_a_frag(lf, t * 16 + m, h, kf);
  __syncthreads();   // lf consumed; reuse as the H (32x256) tile

  // ---- GEMM1: (32x320) @ (320x256) + bias, ReLU -> LDS bf16 ----
  for (int nt = 0; nt < 16; ++nt) {
    v8f acc[MT];
#pragma unroll
    for (int t = 0; t < MT; ++t)
#pragma unroll
      for (int v = 0; v < 8; ++v) acc[t][v] = bias1[nt];
#pragma unroll
    for (int kf = 0; kf < 10; ++kf) {
      const v16bf bfrag =
          *(const v16bf*)(W1b + ((size_t)(kf * 16 + nt) * 32 + lane) * 16);
#pragma unroll
      for (int t = 0; t < MT; ++t)
        acc[t] = __builtin_amdgcn_wmma_f32_16x16x32_bf16(
            false, afrag[t][kf], false, bfrag, (short)0, acc[t], false, false);
    }
#pragma unroll
    for (int t = 0; t < MT; ++t)
#pragma unroll
      for (int v = 0; v < 8; ++v)
        lf[t * 16 + h * 8 + v][nt * 16 + m] = (bf16)fmaxf(acc[t][v], 0.f);
  }
  __syncthreads();

  // ---- A-fragments for GEMM2 (K = 256 -> 8 frags x MT tiles) ----
  v16bf hfrag[MT][8];
#pragma unroll
  for (int t = 0; t < MT; ++t)
#pragma unroll
    for (int kf = 0; kf < 8; ++kf)
      hfrag[t][kf] = load_a_frag(lf, t * 16 + m, h, kf);

  // Destinations for the rows this lane owns in the D fragments.
  int dstRow[MT][8];
#pragma unroll
  for (int t = 0; t < MT; ++t)
#pragma unroll
    for (int v = 0; v < 8; ++v)
      dstRow[t][v] = edge_idx[2 * (e0 + t * 16 + h * 8 + v) + 1];

  // ---- GEMM2: (32x256) @ (256x64) + bias -> scatter-add (f32 atomics) ----
  for (int nt = 0; nt < 4; ++nt) {
    v8f acc[MT];
#pragma unroll
    for (int t = 0; t < MT; ++t)
#pragma unroll
      for (int v = 0; v < 8; ++v) acc[t][v] = bias2[nt];
#pragma unroll
    for (int kf = 0; kf < 8; ++kf) {
      const v16bf bfrag =
          *(const v16bf*)(W2b + ((size_t)(kf * 4 + nt) * 32 + lane) * 16);
#pragma unroll
      for (int t = 0; t < MT; ++t)
        acc[t] = __builtin_amdgcn_wmma_f32_16x16x32_bf16(
            false, hfrag[t][kf], false, bfrag, (short)0, acc[t], false, false);
    }
#pragma unroll
    for (int t = 0; t < MT; ++t)
#pragma unroll
      for (int v = 0; v < 8; ++v)
        unsafeAtomicAdd(
            &agg[((size_t)b * NA + dstRow[t][v]) * EMB + nt * 16 + m],
            acc[t][v]);
  }
}

// ---------------------------------------------------------------------------
// Node kernel: concat(x0, x_last, agg) (160) -> MLP2 (160->256 ReLU -> 64).
// Each wave owns a 32-node tile.
// ---------------------------------------------------------------------------
__global__ __launch_bounds__(32 * WAVES, 1)
void node_kernel(const float* __restrict__ x0,
                 const float* __restrict__ x_last,
                 const float* __restrict__ agg,
                 const float* __restrict__ b1,
                 const float* __restrict__ b2,
                 const bf16*  __restrict__ W3b,
                 const bf16*  __restrict__ W4b,
                 float*       __restrict__ out) {
  __shared__ bf16 smem[WAVES][MROWS][LFPAD];

  const int tid   = threadIdx.x;
  const int lane  = tid & 31;
  const int wave  = tid >> 5;
  const int b     = blockIdx.y;
  const int node0 = (blockIdx.x * WAVES + wave) * MROWS;
  const int m     = lane & 15;
  const int h     = lane >> 4;
  bf16 (&lf)[MROWS][LFPAD] = smem[wave];

  // ---- build concat features (160) for rows m, m+16 ----
#pragma unroll
  for (int r = 0; r < MT; ++r) {
    const int row  = r * 16 + m;
    const int node = node0 + row;
    const float* xp = x0     + ((size_t)b * NA + node) * FEAT;
    const float* lp = x_last + ((size_t)b * NA + node) * EMB;
    const float* ap = agg    + ((size_t)b * NA + node) * EMB;
    for (int f = h * 80; f < h * 80 + 80; ++f) {
      const float val = (f < FEAT) ? xp[f]
                      : (f < FEAT + EMB) ? lp[f - FEAT]
                      : ap[f - FEAT - EMB];
      lf[row][f] = (bf16)val;
    }
  }
  __syncthreads();

  float bias1[16], bias2[4];
#pragma unroll
  for (int nt = 0; nt < 16; ++nt) bias1[nt] = b1[nt * 16 + m];
#pragma unroll
  for (int nt = 0; nt < 4; ++nt)  bias2[nt] = b2[nt * 16 + m];

  v16bf afrag[MT][5];             // K = 160 -> 5 fragments
#pragma unroll
  for (int t = 0; t < MT; ++t)
#pragma unroll
    for (int kf = 0; kf < 5; ++kf)
      afrag[t][kf] = load_a_frag(lf, t * 16 + m, h, kf);
  __syncthreads();

  // ---- GEMM1: (32x160) @ (160x256) + bias, ReLU -> LDS ----
  for (int nt = 0; nt < 16; ++nt) {
    v8f acc[MT];
#pragma unroll
    for (int t = 0; t < MT; ++t)
#pragma unroll
      for (int v = 0; v < 8; ++v) acc[t][v] = bias1[nt];
#pragma unroll
    for (int kf = 0; kf < 5; ++kf) {
      const v16bf bfrag =
          *(const v16bf*)(W3b + ((size_t)(kf * 16 + nt) * 32 + lane) * 16);
#pragma unroll
      for (int t = 0; t < MT; ++t)
        acc[t] = __builtin_amdgcn_wmma_f32_16x16x32_bf16(
            false, afrag[t][kf], false, bfrag, (short)0, acc[t], false, false);
    }
#pragma unroll
    for (int t = 0; t < MT; ++t)
#pragma unroll
      for (int v = 0; v < 8; ++v)
        lf[t * 16 + h * 8 + v][nt * 16 + m] = (bf16)fmaxf(acc[t][v], 0.f);
  }
  __syncthreads();

  v16bf hfrag[MT][8];             // K = 256 -> 8 fragments
#pragma unroll
  for (int t = 0; t < MT; ++t)
#pragma unroll
    for (int kf = 0; kf < 8; ++kf)
      hfrag[t][kf] = load_a_frag(lf, t * 16 + m, h, kf);

  // ---- GEMM2: (32x256) @ (256x64) + bias -> out ----
  for (int nt = 0; nt < 4; ++nt) {
    v8f acc[MT];
#pragma unroll
    for (int t = 0; t < MT; ++t)
#pragma unroll
      for (int v = 0; v < 8; ++v) acc[t][v] = bias2[nt];
#pragma unroll
    for (int kf = 0; kf < 8; ++kf) {
      const v16bf bfrag =
          *(const v16bf*)(W4b + ((size_t)(kf * 4 + nt) * 32 + lane) * 16);
#pragma unroll
      for (int t = 0; t < MT; ++t)
        acc[t] = __builtin_amdgcn_wmma_f32_16x16x32_bf16(
            false, hfrag[t][kf], false, bfrag, (short)0, acc[t], false, false);
    }
#pragma unroll
    for (int t = 0; t < MT; ++t)
#pragma unroll
      for (int v = 0; v < 8; ++v)
        out[((size_t)b * NA + node0 + t * 16 + h * 8 + v) * EMB + nt * 16 + m] =
            acc[t][v];
  }
}

// ---------------------------------------------------------------------------
// Launch
// ---------------------------------------------------------------------------
extern "C" void kernel_launch(void* const* d_in, const int* in_sizes, int n_in,
                              void* d_out, int out_size, void* d_ws, size_t ws_size,
                              hipStream_t stream) {
  (void)in_sizes; (void)n_in; (void)out_size; (void)ws_size;
  const float* x0        = (const float*)d_in[0];
  const float* x_last    = (const float*)d_in[1];
  const float* edge_attr = (const float*)d_in[2];
  const float* conv_w    = (const float*)d_in[3];
  const float* conv_b    = (const float*)d_in[4];
  const float* mlp1_w1   = (const float*)d_in[5];
  const float* mlp1_b1   = (const float*)d_in[6];
  const float* mlp1_w2   = (const float*)d_in[7];
  const float* mlp1_b2   = (const float*)d_in[8];
  const float* mlp2_w1   = (const float*)d_in[9];
  const float* mlp2_b1   = (const float*)d_in[10];
  const float* mlp2_w2   = (const float*)d_in[11];
  const float* mlp2_b2   = (const float*)d_in[12];
  const int*   edge_idx  = (const int*)d_in[13];
  float* out = (float*)d_out;

  // Workspace layout (all 256B-aligned): agg then 4 packed bf16 weight blocks.
  char* ws = (char*)d_ws;
  float* agg = (float*)ws;
  size_t off = (size_t)BS * NA * EMB * sizeof(float);          // 1 MB
  bf16* W1b = (bf16*)(ws + off); off += (size_t)MLP1_IN * H1 * sizeof(bf16);
  bf16* W2b = (bf16*)(ws + off); off += (size_t)H1 * EMB * sizeof(bf16);
  bf16* W3b = (bf16*)(ws + off); off += (size_t)MLP2_IN * H1 * sizeof(bf16);
  bf16* W4b = (bf16*)(ws + off);

  zero_kernel<<<(BS * NA * EMB + 255) / 256, 256, 0, stream>>>(agg, BS * NA * EMB);
  pack_b_kernel<<<(MLP1_IN * H1) / 256, 256, 0, stream>>>(mlp1_w1, W1b, MLP1_IN, H1);
  pack_b_kernel<<<(H1 * EMB) / 256,   256, 0, stream>>>(mlp1_w2, W2b, H1, EMB);
  pack_b_kernel<<<(MLP2_IN * H1) / 256, 256, 0, stream>>>(mlp2_w1, W3b, MLP2_IN, H1);
  pack_b_kernel<<<(H1 * EMB) / 256,   256, 0, stream>>>(mlp2_w2, W4b, H1, EMB);

  edge_kernel<<<dim3(NE / (MROWS * WAVES), BS), 32 * WAVES, 0, stream>>>(
      x0, x_last, edge_attr, conv_w, conv_b, mlp1_b1, mlp1_b2, edge_idx,
      W1b, W2b, agg);
  node_kernel<<<dim3(NA / (MROWS * WAVES), BS), 32 * WAVES, 0, stream>>>(
      x0, x_last, agg, mlp2_b1, mlp2_b2, W3b, W4b, out);
}
